// LinearTemporalAttention_17875653886364
// MI455X (gfx1250) — matrix operand order repaired and checked by reading, not compiled, not run on hardware
//
#include <hip/hip_runtime.h>
#include <hip/hip_bf16.h>

// ---------------------------------------------------------------------------
// LinearTemporalAttention block for MI455X (gfx1250), bf16 WMMA pipeline
// with async global->LDS staging (ASYNCcnt path).
//   B=4, T=4096, C=1024, H=16, D=64
// ---------------------------------------------------------------------------

#define B_ 4
#define T_ 4096
#define C_ 1024
#define H_ 16
#define D_ 64
#define BT_ (B_ * T_)           // 16384 rows
#define BTC_ ((size_t)BT_ * C_) // 16,777,216 elements

typedef __attribute__((ext_vector_type(16))) __bf16        v16bf;
typedef __attribute__((ext_vector_type(8)))  float         v8f;
typedef __attribute__((ext_vector_type(4)))  unsigned int  u32x4;

union FragAB {
    v16bf v;
    struct { u32x4 lo, hi; } u;
};

__device__ __forceinline__ unsigned short f32_to_bf16_rne(float f) {
    union { float f; unsigned int u; } c; c.f = f;
    unsigned int u = c.u;
    u += 0x7fffu + ((u >> 16) & 1u);          // round to nearest even
    return (unsigned short)(u >> 16);
}
__device__ __forceinline__ float bf16u_to_f32(unsigned short h) {
    union { unsigned int u; float f; } c; c.u = ((unsigned int)h) << 16;
    return c.f;
}

// async global->LDS 16-byte copy (gfx1250, tracked with ASYNCcnt)
__device__ __forceinline__ void async_copy_b128(const void* gsrc, void* lds_dst) {
    // flat shared address: low 32 bits are the LDS byte offset (aperture map)
    unsigned lds_addr = (unsigned)(uintptr_t)lds_dst;
    asm volatile("global_load_async_to_lds_b128 %0, %1, off"
                 :: "v"(lds_addr), "v"(gsrc) : "memory");
}
__device__ __forceinline__ void wait_async0() {
    asm volatile("s_wait_asynccnt 0x0" ::: "memory");
}

// ---------------------------------------------------------------------------
// 1) fp32 -> bf16 convert (for x)
// ---------------------------------------------------------------------------
__global__ __launch_bounds__(256) void cvt_f32_bf16_kernel(
    const float* __restrict__ src, unsigned short* __restrict__ dst, int n) {
    int i = blockIdx.x * 256 + threadIdx.x;
    if (i < n) dst[i] = f32_to_bf16_rne(src[i]);
}

// ---------------------------------------------------------------------------
// 2) fp32 [K,N] -> bf16 [N,K] transpose-convert (for weights)
// ---------------------------------------------------------------------------
__global__ __launch_bounds__(256) void transpose_cvt_kernel(
    const float* __restrict__ src, unsigned short* __restrict__ dst,
    int R, int Cc) {
    int idx = blockIdx.x * 256 + threadIdx.x;
    if (idx >= R * Cc) return;
    int r = idx / Cc, c = idx % Cc;
    dst[(size_t)c * R + r] = f32_to_bf16_rne(src[idx]);
}

// ---------------------------------------------------------------------------
// 3) bf16 WMMA GEMM:  out[M,N] = A[M,K] @ Bt[N,K]^T  (+ epilogue)
//    mode 0: out_bf16 = acc + bias
//    mode 1: out_bf16 = relu(acc + bias) + 1
//    mode 2: out_f32  = acc + bias + resid            (final projection)
//    Block tile 128x128, K-step 32, 8 waves (wave32), each wave 64x32.
//    Staging: global_load_async_to_lds_b128, double buffered.
// ---------------------------------------------------------------------------
#define BM 128
#define BN 128
#define BK 32
#define LDT 40   // padded row stride in bf16 elems: 80 B = 5*16 B (aligned)

__device__ __forceinline__ void stage_tile_async(
    const unsigned short* __restrict__ g, int gRowBase, int K, int k0,
    unsigned short (*lds)[LDT], int tid) {
#pragma unroll
    for (int i = 0; i < 2; ++i) {
        int c = tid + 256 * i;                 // 0..511 chunks of 8 bf16
        int row  = c >> 2;                     // 0..127
        int col8 = (c & 3) << 3;               // 0,8,16,24
        async_copy_b128(g + (size_t)(gRowBase + row) * K + k0 + col8,
                        &lds[row][col8]);
    }
}

__global__ __launch_bounds__(256) void gemm_bf16_wmma_kernel(
    const unsigned short* __restrict__ Ag,   // [M,K] bf16
    const unsigned short* __restrict__ Btg,  // [N,K] bf16 (W transposed)
    const float* __restrict__ bias,          // [N]
    const float* __restrict__ resid,         // [M,N] fp32 or nullptr
    unsigned short* __restrict__ outb,       // [M,N] bf16 (modes 0,1)
    float* __restrict__ outf,                // [M,N] f32  (mode 2)
    int M, int N, int K, int mode) {
    __shared__ __align__(16) unsigned short As[2][BM][LDT];
    __shared__ __align__(16) unsigned short Bs[2][BN][LDT];

    const int tid    = threadIdx.x;
    const int wave   = tid >> 5;       // 0..7
    const int lane   = tid & 31;
    const int laneLo = lane & 15;
    const int laneHi = lane >> 4;      // 0 or 1

    const int bm = blockIdx.y * BM;
    const int bn = blockIdx.x * BN;
    const int wm = (wave >> 2) * 64;   // 0 / 64
    const int wn = (wave & 3) * 32;    // 0,32,64,96

    v8f acc[4][2];
#pragma unroll
    for (int mt = 0; mt < 4; ++mt)
#pragma unroll
        for (int nt = 0; nt < 2; ++nt) acc[mt][nt] = {};

    const int kTiles = K / BK;
    stage_tile_async(Ag,  bm, K, 0, As[0], tid);
    stage_tile_async(Btg, bn, K, 0, Bs[0], tid);
    wait_async0();
    __syncthreads();

    for (int kt = 0; kt < kTiles; ++kt) {
        const int cur = kt & 1;
        if (kt + 1 < kTiles) {
            stage_tile_async(Ag,  bm, K, (kt + 1) * BK, As[cur ^ 1], tid);
            stage_tile_async(Btg, bn, K, (kt + 1) * BK, Bs[cur ^ 1], tid);
        }
        if (kt + 2 < kTiles) {   // gfx1250 global_prefetch_b8 two stages ahead
            int kn = (kt + 2) * BK;
            __builtin_prefetch(Ag  + (size_t)(bm + (tid >> 2)) * K + kn + ((tid & 3) << 3), 0, 1);
            __builtin_prefetch(Btg + (size_t)(bn + (tid >> 2)) * K + kn + ((tid & 3) << 3), 0, 1);
        }

        v16bf af[4], bfrag[2];
#pragma unroll
        for (int mt = 0; mt < 4; ++mt) {
            const int row = wm + mt * 16 + laneLo;
            FragAB a;                                   // A 16x32: K chunks of 8
            a.u.lo = *(const u32x4*)&As[cur][row][laneHi * 8];
            a.u.hi = *(const u32x4*)&As[cur][row][16 + laneHi * 8];
            af[mt] = a.v;
        }
#pragma unroll
        for (int nt = 0; nt < 2; ++nt) {
            const int col = wn + nt * 16 + laneLo;
            FragAB b;                                   // B 32x16: K chunks of 8
            b.u.lo = *(const u32x4*)&Bs[cur][col][laneHi * 8];
            b.u.hi = *(const u32x4*)&Bs[cur][col][16 + laneHi * 8];
            bfrag[nt] = b.v;
        }
#pragma unroll
        for (int mt = 0; mt < 4; ++mt)
#pragma unroll
            for (int nt = 0; nt < 2; ++nt)
                acc[mt][nt] = __builtin_amdgcn_wmma_f32_16x16x32_bf16(
                    false, af[mt], false, bfrag[nt], (short)0, acc[mt][nt],
                    false, false);

        wait_async0();      // next-stage async copies landed in LDS
        __syncthreads();
    }

    // epilogue
#pragma unroll
    for (int mt = 0; mt < 4; ++mt) {
#pragma unroll
        for (int nt = 0; nt < 2; ++nt) {
            const int col = bn + wn + nt * 16 + laneLo;
            const float bcol = bias[col];
#pragma unroll
            for (int r = 0; r < 8; ++r) {
                const int row = bm + wm + mt * 16 + laneHi * 8 + r;
                float v = acc[mt][nt][r] + bcol;
                if (mode == 1) v = fmaxf(v, 0.0f) + 1.0f;
                const size_t off = (size_t)row * N + col;
                if (mode == 2) outf[off] = v + resid[off];
                else           outb[off] = f32_to_bf16_rne(v);
            }
        }
    }
}

// ---------------------------------------------------------------------------
// 4) per-(b,h): kv[d][e] = sum_t k[t,d]*v[t,e];  ksum[d] = sum_t k[t,d]
// ---------------------------------------------------------------------------
__global__ __launch_bounds__(256) void kv_ksum_kernel(
    const unsigned short* __restrict__ kb, const unsigned short* __restrict__ vb,
    float* __restrict__ kv, float* __restrict__ ksum) {
    __shared__ __align__(16) unsigned short kt[64][64];
    __shared__ __align__(16) unsigned short vt[64][64];
    const int bh = blockIdx.x;             // b*H + h
    const int b = bh >> 4, h = bh & 15;
    const size_t headoff = (size_t)b * T_ * C_ + (size_t)h * D_;
    const int tid = threadIdx.x;
    const int d  = tid >> 2;               // 0..63
    const int eb = (tid & 3) << 4;         // 0,16,32,48

    float acc[16];
#pragma unroll
    for (int j = 0; j < 16; ++j) acc[j] = 0.0f;
    float ks = 0.0f;

    for (int t0 = 0; t0 < T_; t0 += 64) {
#pragma unroll
        for (int i = 0; i < 2; ++i) {
            int c = tid + 256 * i;         // 0..511 chunks of 8
            int tr = c >> 3, d8 = (c & 7) << 3;
            size_t g = headoff + (size_t)(t0 + tr) * C_ + d8;
            *(u32x4*)&kt[tr][d8] = *(const u32x4*)(kb + g);
            *(u32x4*)&vt[tr][d8] = *(const u32x4*)(vb + g);
        }
        __syncthreads();
#pragma unroll 8
        for (int tt = 0; tt < 64; ++tt) {
            const float kd = bf16u_to_f32(kt[tt][d]);
#pragma unroll
            for (int j = 0; j < 16; ++j)
                acc[j] = fmaf(kd, bf16u_to_f32(vt[tt][eb + j]), acc[j]);
        }
        if (tid < 64) {
#pragma unroll 8
            for (int tt = 0; tt < 64; ++tt) ks += bf16u_to_f32(kt[tt][tid]);
        }
        __syncthreads();
    }
    float* kvp = kv + (size_t)bh * D_ * D_ + (size_t)d * D_ + eb;
#pragma unroll
    for (int j = 0; j < 16; ++j) kvp[j] = acc[j];
    if (tid < 64) ksum[bh * D_ + tid] = ks;
}

// ---------------------------------------------------------------------------
// 5) att[b,t,h,e] = (sum_d q*kv[d,e]) / (sum_d q*ksum[d] + eps)  -> bf16
// ---------------------------------------------------------------------------
__global__ __launch_bounds__(256) void attn_out_kernel(
    const unsigned short* __restrict__ qb, const float* __restrict__ kv,
    const float* __restrict__ ksum, unsigned short* __restrict__ att) {
    const size_t idx = (size_t)blockIdx.x * 256 + threadIdx.x; // over B*T*C
    const int e = (int)(idx & 63);
    const int h = (int)((idx >> 6) & 15);
    const size_t bt = idx >> 10;           // 0..16383
    const int b = (int)(bt >> 12);         // / T_
    const int bh = b * H_ + h;

    const unsigned short* qrow = qb + bt * C_ + (size_t)h * D_;
    const float* kvp = kv + (size_t)bh * D_ * D_;
    const float* ksp = ksum + (size_t)bh * D_;

    float acc = 0.0f, z = 0.0f;
#pragma unroll 16
    for (int d = 0; d < D_; ++d) {
        const float qd = bf16u_to_f32(qrow[d]);
        acc = fmaf(qd, kvp[d * D_ + e], acc);
        z   = fmaf(qd, ksp[d], z);
    }
    att[idx] = f32_to_bf16_rne(acc / (z + 1e-6f));
}

// ---------------------------------------------------------------------------
// 6) in-place LayerNorm over last dim (C=1024), one row per block
// ---------------------------------------------------------------------------
__global__ __launch_bounds__(256) void layernorm_kernel(
    float* __restrict__ y, const float* __restrict__ lw,
    const float* __restrict__ lb) {
    __shared__ float s1[256], s2[256];
    const int tid = threadIdx.x;
    float* p = y + (size_t)blockIdx.x * C_;
    float vals[4];
    float a = 0.0f, b = 0.0f;
#pragma unroll
    for (int i = 0; i < 4; ++i) {
        float x = p[tid + 256 * i];
        vals[i] = x; a += x; b += x * x;
    }
    s1[tid] = a; s2[tid] = b;
    __syncthreads();
    for (int st = 128; st > 0; st >>= 1) {
        if (tid < st) { s1[tid] += s1[tid + st]; s2[tid] += s2[tid + st]; }
        __syncthreads();
    }
    const float mu  = s1[0] * (1.0f / C_);
    const float var = s2[0] * (1.0f / C_) - mu * mu;
    const float rs  = rsqrtf(var + 1e-5f);
#pragma unroll
    for (int i = 0; i < 4; ++i) {
        int c = tid + 256 * i;
        p[c] = (vals[i] - mu) * rs * lw[c] + lb[c];
    }
}

// ---------------------------------------------------------------------------
// launch
// ---------------------------------------------------------------------------
extern "C" void kernel_launch(void* const* d_in, const int* in_sizes, int n_in,
                              void* d_out, int out_size, void* d_ws, size_t ws_size,
                              hipStream_t stream) {
    (void)in_sizes; (void)n_in; (void)out_size; (void)ws_size;
    const float* x    = (const float*)d_in[0];
    const float* Wq   = (const float*)d_in[1];
    const float* bq   = (const float*)d_in[2];
    const float* Wk   = (const float*)d_in[3];
    const float* bk   = (const float*)d_in[4];
    const float* Wv   = (const float*)d_in[5];
    const float* bv   = (const float*)d_in[6];
    const float* Wo   = (const float*)d_in[7];
    const float* bo   = (const float*)d_in[8];
    const float* ln_w = (const float*)d_in[9];
    const float* ln_b = (const float*)d_in[10];
    float* out = (float*)d_out;

    // workspace carve (256-B aligned)
    char* p = (char*)d_ws;
    auto carve = [&](size_t bytes) {
        char* r = p;
        p += (bytes + 255) & ~(size_t)255;
        return r;
    };
    unsigned short* x_bf = (unsigned short*)carve(BTC_ * 2);  // also reused as att
    unsigned short* WqT  = (unsigned short*)carve((size_t)C_ * C_ * 2);
    unsigned short* WkT  = (unsigned short*)carve((size_t)C_ * C_ * 2);
    unsigned short* WvT  = (unsigned short*)carve((size_t)C_ * C_ * 2);
    unsigned short* WoT  = (unsigned short*)carve((size_t)C_ * C_ * 2);
    unsigned short* qb   = (unsigned short*)carve(BTC_ * 2);
    unsigned short* kb   = (unsigned short*)carve(BTC_ * 2);
    unsigned short* vb   = (unsigned short*)carve(BTC_ * 2);
    float* kvbuf = (float*)carve((size_t)B_ * H_ * D_ * D_ * 4);
    float* ksum  = (float*)carve((size_t)B_ * H_ * D_ * 4);
    unsigned short* att = x_bf;   // x_bf dead after QKV GEMMs

    // 1) convert x to bf16
    cvt_f32_bf16_kernel<<<(int)(BTC_ / 256), 256, 0, stream>>>(x, x_bf, (int)BTC_);
    // 2) transpose+convert weights
    const int wgrid = (C_ * C_) / 256;
    transpose_cvt_kernel<<<wgrid, 256, 0, stream>>>(Wq, WqT, C_, C_);
    transpose_cvt_kernel<<<wgrid, 256, 0, stream>>>(Wk, WkT, C_, C_);
    transpose_cvt_kernel<<<wgrid, 256, 0, stream>>>(Wv, WvT, C_, C_);
    transpose_cvt_kernel<<<wgrid, 256, 0, stream>>>(Wo, WoT, C_, C_);

    // 3) Q/K/V projections (bf16 WMMA)
    dim3 gg(C_ / BN, BT_ / BM);   // (8, 128)
    gemm_bf16_wmma_kernel<<<gg, 256, 0, stream>>>(x_bf, WqT, bq, nullptr, qb, nullptr, BT_, C_, C_, 1);
    gemm_bf16_wmma_kernel<<<gg, 256, 0, stream>>>(x_bf, WkT, bk, nullptr, kb, nullptr, BT_, C_, C_, 1);
    gemm_bf16_wmma_kernel<<<gg, 256, 0, stream>>>(x_bf, WvT, bv, nullptr, vb, nullptr, BT_, C_, C_, 0);

    // 4) kv / ksum reduction per (b,h)
    kv_ksum_kernel<<<B_ * H_, 256, 0, stream>>>(kb, vb, kvbuf, ksum);

    // 5) attention output (bf16) — overwrites x_bf
    attn_out_kernel<<<(int)(BTC_ / 256), 256, 0, stream>>>(qb, kvbuf, ksum, att);

    // 6) final projection + bias + residual -> d_out (fp32)
    gemm_bf16_wmma_kernel<<<gg, 256, 0, stream>>>(att, WoT, bo, x, nullptr, out, BT_, C_, C_, 2);

    // 7) in-place LayerNorm
    layernorm_kernel<<<BT_, 256, 0, stream>>>(out, ln_w, ln_b);
}
